// GraphEncoder_89550068121929
// MI455X (gfx1250) — matrix operand order repaired and checked by reading, not compiled, not run on hardware
//
#include <hip/hip_runtime.h>
#include <hip/hip_bf16.h>
#include <math.h>

// ---------------- problem constants ----------------
#define N_NODES 50000
#define N_EDGES 400000
#define N_GRAPH 100
#define HID     128
#define EMBD    64
#define HSTR    72      // h row: 69 valid + 3 pad (bf16)
#define K1R     164     // edge gemm1 real K
#define K1P     192     // padded to 6 chunks of 32
#define KNR     197     // node gemm1 real K
#define KNP     224     // padded to 7 chunks of 32
#define ZD      256
#define KZ      384     // head gemm K (HID*3), 12 chunks
#define MP      112     // graphs padded to 7 row-tiles

typedef __bf16 bf16;
typedef __attribute__((ext_vector_type(16))) __bf16 v16bf;
typedef __attribute__((ext_vector_type(8)))  __bf16 v8bf;
typedef __attribute__((ext_vector_type(8)))  float  v8f;

// Build a 16x32 A-fragment / 32x16 B-fragment: two 16B halves.
__device__ __forceinline__ v16bf frag2(const bf16* p0, const bf16* p1) {
  union { v16bf v; v8bf h[2]; } u;
  u.h[0] = *(const v8bf*)p0;
  u.h[1] = *(const v8bf*)p1;
  return u.v;
}

__device__ __forceinline__ v8f wmma_bf16(v16bf a, v16bf b, v8f c) {
  return __builtin_amdgcn_wmma_f32_16x16x32_bf16(false, a, false, b, (short)0, c,
                                                 false, false);
}

// Async memory->LDS staging (CDNA5 GLOBAL_LOAD_ASYNC_TO_LDS, ASYNCcnt-tracked).
// GVS mode: LDS address comes from the vdst VGPR; memory address is SGPR base plus VGPR offset.
// Low 32 bits of a flat LDS pointer are the wave-relative LDS byte offset.
__device__ __forceinline__ void async_stage_b128(void* lds_dst, const void* gsrc,
                                                 int bytes, int tid) {
  const unsigned lds_base = (unsigned)(size_t)lds_dst;
  for (int o = tid * 16; o < bytes; o += 256 * 16) {
    const unsigned ldsa = lds_base + (unsigned)o;
    const unsigned goff = (unsigned)o;
    asm volatile("global_load_async_to_lds_b128 %0, %1, %2"
                 :: "v"(ldsa), "v"(goff), "s"(gsrc)
                 : "memory");
  }
}
__device__ __forceinline__ void async_wait_all() {
  asm volatile("s_wait_asynccnt 0x0" ::: "memory");
}

// One wave computes a 16-row strip x 4 col-tiles (64 cols) of C = A @ W^T'
//   A : LDS, row-major [rows][KP] bf16
//   W : LDS, W^T row-major [128][KP] bf16 (lane's B frag = two 16B runs, 32B apart)
// A-frag (16-bit A 16x32): lane<16 -> K 0-7 & 16-23; lane>=16 -> K 8-15 & 24-31.
// B-frag (16-bit B 32x16): lane<16 -> col N=lane, K 0-7 & 16-23; lane>=16 -> K 8-15 & 24-31.
// Software-pipelined: fragments for chunk kc+1 are loaded while chunk kc's
// 4 WMMAs execute, so ds_load latency overlaps the matrix pipe.
template <int KP, int KCH>
__device__ __forceinline__ void gemm_tile(const bf16* __restrict__ A,
                                          const bf16* __restrict__ W,
                                          int lane, int rt, int cg, v8f acc[4]) {
  const int arow  = rt * 16 + (lane & 15);
  const int khalf = (lane >> 4) * 8;
  const int bncol = lane & 15;
  const int bkh   = (lane >> 4) * 8;
  const bf16* aptr = &A[arow * KP + khalf];
  const bf16* b0p = &W[((cg * 4 + 0) * 16 + bncol) * KP + bkh];
  const bf16* b1p = &W[((cg * 4 + 1) * 16 + bncol) * KP + bkh];
  const bf16* b2p = &W[((cg * 4 + 2) * 16 + bncol) * KP + bkh];
  const bf16* b3p = &W[((cg * 4 + 3) * 16 + bncol) * KP + bkh];
#pragma unroll
  for (int t = 0; t < 4; ++t) acc[t] = (v8f)0.0f;

  v16bf a  = frag2(aptr, aptr + 16);
  v16bf b0 = frag2(b0p, b0p + 16);
  v16bf b1 = frag2(b1p, b1p + 16);
  v16bf b2 = frag2(b2p, b2p + 16);
  v16bf b3 = frag2(b3p, b3p + 16);
#pragma unroll
  for (int kc = 0; kc < KCH; ++kc) {
    v16bf an = a, c0 = b0, c1 = b1, c2 = b2, c3 = b3;
    if (kc + 1 < KCH) {               // prefetch next chunk's fragments
      const int o = (kc + 1) * 32;
      an = frag2(aptr + o, aptr + o + 16);
      c0 = frag2(b0p + o, b0p + o + 16);
      c1 = frag2(b1p + o, b1p + o + 16);
      c2 = frag2(b2p + o, b2p + o + 16);
      c3 = frag2(b3p + o, b3p + o + 16);
    }
    acc[0] = wmma_bf16(a, b0, acc[0]);
    acc[1] = wmma_bf16(a, b1, acc[1]);
    acc[2] = wmma_bf16(a, b2, acc[2]);
    acc[3] = wmma_bf16(a, b3, acc[3]);
    a = an; b0 = c0; b1 = c1; b2 = c2; b3 = c3;
  }
}

// ---------------- kernel 1: zero accumulators ----------------
__global__ void zero_ws(float* __restrict__ msg_sum, float* __restrict__ cnt,
                        float* __restrict__ pooled) {
  const size_t i0 = (size_t)blockIdx.x * blockDim.x + threadIdx.x;
  const size_t st = (size_t)gridDim.x * blockDim.x;
  for (size_t k = i0; k < (size_t)N_NODES * HID; k += st) msg_sum[k] = 0.0f;
  for (size_t k = i0; k < (size_t)N_NODES; k += st) cnt[k] = 0.0f;
  for (size_t k = i0; k < (size_t)MP * KZ; k += st) pooled[k] = 0.0f;
}

// ---------------- kernel 2: weight prep (transpose->bf16, fold BN) ----------------
struct PrepArgs {
  const float *W1e, *b1e, *g1e, *be1e, *m1e, *v1e;
  const float *W2e, *b2e, *g2e, *be2e, *m2e, *v2e;
  const float *W1n, *b1n, *g1n, *be1n, *m1n, *v1n;
  const float *W2n, *b2n, *g2n, *be2n, *m2n, *v2n;
  const float *Wm,  *bmu, *gmu, *bemu, *mmu, *vmu;
  const float *Ws,  *bsg, *gsg, *besg, *msig, *vsig;
  bf16 *W1eT, *W2eT, *W1nT, *W2nT, *WmT, *WsT;
  float *affE, *affN, *affMu, *affSg;   // [scale1,shift1,(scale2,shift2)]
};

__global__ void prep_weights(PrepArgs a) {
  const int i0 = blockIdx.x * blockDim.x + threadIdx.x;
  const int st = gridDim.x * blockDim.x;
  for (int i = i0; i < HID * K1P; i += st) {
    int n = i / K1P, k = i % K1P;
    a.W1eT[i] = (k < K1R) ? (bf16)a.W1e[k * HID + n] : (bf16)0.0f;
  }
  for (int i = i0; i < HID * KNP; i += st) {
    int n = i / KNP, k = i % KNP;
    a.W1nT[i] = (k < KNR) ? (bf16)a.W1n[k * HID + n] : (bf16)0.0f;
  }
  for (int i = i0; i < HID * HID; i += st) {
    int n = i / HID, k = i % HID;
    a.W2eT[i] = (bf16)a.W2e[k * HID + n];
    a.W2nT[i] = (bf16)a.W2n[k * HID + n];
  }
  for (int i = i0; i < ZD * KZ; i += st) {
    int n = i / KZ, k = i % KZ;
    a.WmT[i] = (bf16)a.Wm[k * ZD + n];
    a.WsT[i] = (bf16)a.Ws[k * ZD + n];
  }
  for (int i = i0; i < HID; i += st) {
    float s1 = rsqrtf(a.v1e[i] + 1e-5f) * a.g1e[i];
    a.affE[i]            = s1;
    a.affE[HID + i]      = (a.b1e[i] - a.m1e[i]) * s1 + a.be1e[i];
    float s2 = rsqrtf(a.v2e[i] + 1e-5f) * a.g2e[i];
    a.affE[2 * HID + i]  = s2;
    a.affE[3 * HID + i]  = (a.b2e[i] - a.m2e[i]) * s2 + a.be2e[i];
    float t1 = rsqrtf(a.v1n[i] + 1e-5f) * a.g1n[i];
    a.affN[i]            = t1;
    a.affN[HID + i]      = (a.b1n[i] - a.m1n[i]) * t1 + a.be1n[i];
    float t2 = rsqrtf(a.v2n[i] + 1e-5f) * a.g2n[i];
    a.affN[2 * HID + i]  = t2;
    a.affN[3 * HID + i]  = (a.b2n[i] - a.m2n[i]) * t2 + a.be2n[i];
  }
  for (int i = i0; i < ZD; i += st) {
    float sm = rsqrtf(a.vmu[i] + 1e-5f) * a.gmu[i];
    a.affMu[i]      = sm;
    a.affMu[ZD + i] = (a.bmu[i] - a.mmu[i]) * sm + a.bemu[i];
    float ss = rsqrtf(a.vsig[i] + 1e-5f) * a.gsg[i];
    a.affSg[i]      = ss;
    a.affSg[ZD + i] = (a.bsg[i] - a.msig[i]) * ss + a.besg[i];
  }
}

// ---------------- kernel 3: atom encoder -> h bf16 [N][72] ----------------
struct EmbPtrs { const float* p[16]; };

__global__ __launch_bounds__(288) void atom_encode(
    EmbPtrs emb, const int* __restrict__ node_cat,
    const float* __restrict__ mu_r, bf16* __restrict__ hbf) {
  const int n = blockIdx.x * 4 + threadIdx.x / 72;
  const int d = threadIdx.x % 72;
  if (n >= N_NODES) return;
  bf16* out = &hbf[(size_t)n * HSTR + d];
  if (d < EMBD) {
    float s = 0.0f;
#pragma unroll
    for (int i = 0; i < 16; ++i) {
      const int c = node_cat[(size_t)n * 16 + i];
      s += emb.p[i][(size_t)c * EMBD + d];
    }
    *out = (bf16)s;
  } else if (d < 69) {
    *out = (bf16)__logf(mu_r[(size_t)n * 5 + (d - 64)]);
  } else {
    *out = (bf16)0.0f;   // pad
  }
}

// ---------------- kernel 4: fused edge MLP + scatter (persistent) ----------------
__global__ __launch_bounds__(256) void edge_mlp_scatter(
    const bf16* __restrict__ hbf, const float* __restrict__ coords,
    const float* __restrict__ edge_feat, const int* __restrict__ src,
    const int* __restrict__ dst, const bf16* __restrict__ W1T,
    const bf16* __restrict__ W2T, const float* __restrict__ aff,
    float* __restrict__ msg_sum, float* __restrict__ cnt) {
  __shared__ bf16  sA1[64 * K1P];
  __shared__ bf16  sW1[HID * K1P];
  __shared__ bf16  sA2[64 * HID];
  __shared__ bf16  sW2[HID * HID];
  __shared__ float sAff[4 * HID];
  __shared__ int   sDst[64];

  const int tid = threadIdx.x;
  // weights: async memory->LDS DMA, no VGPR round-trip
  async_stage_b128(sW1, W1T, HID * K1P * 2, tid);
  async_stage_b128(sW2, W2T, HID * HID * 2, tid);
  for (int i = tid; i < 4 * HID; i += 256) sAff[i] = aff[i];
  async_wait_all();

  const int lane = tid & 31, wave = tid >> 5;
  const int rt = wave & 3, cg = wave >> 2;
  const int mbase = rt * 16 + 8 * (lane >> 4);
  const int bn = lane & 15;
  const int numTiles = (N_EDGES + 63) / 64;

  for (int tile = blockIdx.x; tile < numTiles; tile += gridDim.x) {
    const int e0 = tile * 64;
    if (tid == 0 && tile + gridDim.x < numTiles) {
      __builtin_prefetch(&src[(size_t)(tile + gridDim.x) * 64], 0, 1);
      __builtin_prefetch(&dst[(size_t)(tile + gridDim.x) * 64], 0, 1);
    }
    {  // assemble A row = [h[src](69) | h[dst](69) | ef(16) | rbf(10) | 0...]
      const int r = tid >> 2, q = tid & 3;
      const int e = e0 + r;
      bf16* row = &sA1[r * K1P];
      if (e < N_EDGES) {
        const int s = src[e], d = dst[e];
        const bf16* hs = &hbf[(size_t)s * HSTR];
        const bf16* hd = &hbf[(size_t)d * HSTR];
        const float* ef = &edge_feat[(size_t)e * 16];
        if (q == 0) {
          sDst[r] = d;
          for (int k = 0; k < 48; ++k) row[k] = hs[k];
        } else if (q == 1) {
          for (int k = 48; k < 69; ++k) row[k] = hs[k];
          for (int k = 0; k < 27; ++k) row[69 + k] = hd[k];
        } else if (q == 2) {
          for (int k = 27; k < 69; ++k) row[69 + k] = hd[k];
          for (int k = 0; k < 6; ++k) row[138 + k] = (bf16)ef[k];
        } else {
          for (int k = 6; k < 16; ++k) row[138 + k] = (bf16)ef[k];
          float dx = coords[s * 3 + 0] - coords[d * 3 + 0];
          float dy = coords[s * 3 + 1] - coords[d * 3 + 1];
          float dz = coords[s * 3 + 2] - coords[d * 3 + 2];
          float d2 = dx * dx + dy * dy + dz * dz;
          float sg = 1.0f;
#pragma unroll
          for (int k = 0; k < 10; ++k) { row[154 + k] = (bf16)__expf(-d2 / sg); sg *= 1.5f; }
#pragma unroll
          for (int k = K1R; k < K1P; ++k) row[k] = (bf16)0.0f;
        }
      } else {
        if (q == 0) sDst[r] = 0;
        for (int k = q * 48; k < q * 48 + 48; ++k) row[k] = (bf16)0.0f;
      }
    }
    __syncthreads();
    if (tid < 64 && (e0 + tid) < N_EDGES) atomicAdd(&cnt[sDst[tid]], 1.0f);

    v8f acc[4];
    gemm_tile<K1P, K1P / 32>(sA1, sW1, lane, rt, cg, acc);

    // epilogue 1: fused bias+BN then LeakyReLU, stage as bf16 A2
#pragma unroll
    for (int t = 0; t < 4; ++t) {
      const int col = (cg * 4 + t) * 16 + bn;
      const float s1 = sAff[col], c1 = sAff[HID + col];
#pragma unroll
      for (int i = 0; i < 8; ++i) {
        float v = acc[t][i] * s1 + c1;
        v = v > 0.0f ? v : 0.01f * v;
        sA2[(mbase + i) * HID + col] = (bf16)v;
      }
    }
    __syncthreads();

    gemm_tile<HID, HID / 32>(sA2, sW2, lane, rt, cg, acc);

    // epilogue 2: fused bias+BN, scatter-add per destination node
#pragma unroll
    for (int t = 0; t < 4; ++t) {
      const int col = (cg * 4 + t) * 16 + bn;
      const float s2 = sAff[2 * HID + col], c2 = sAff[3 * HID + col];
#pragma unroll
      for (int i = 0; i < 8; ++i) {
        const int r = mbase + i;
        if (e0 + r < N_EDGES) {
          atomicAdd(&msg_sum[(size_t)sDst[r] * HID + col], acc[t][i] * s2 + c2);
        }
      }
    }
    __syncthreads();   // protect sA1/sDst before next tile
  }
}

// ---------------- kernel 5: fused node MLP + coord-outer pooling ----------------
__global__ __launch_bounds__(256) void node_mlp_pool(
    const bf16* __restrict__ hbf, const float* __restrict__ msg_sum,
    const float* __restrict__ cnt, const float* __restrict__ coords,
    const int* __restrict__ graph_id, const bf16* __restrict__ W1T,
    const bf16* __restrict__ W2T, const float* __restrict__ aff,
    float* __restrict__ pooled) {
  __shared__ bf16  sA1[64 * KNP];
  __shared__ bf16  sW1[HID * KNP];
  __shared__ bf16  sA2[64 * HID];
  __shared__ bf16  sW2[HID * HID];
  __shared__ float sAff[4 * HID];
  __shared__ float sC[64 * 3];
  __shared__ int   sG[64];

  const int tid = threadIdx.x;
  async_stage_b128(sW1, W1T, HID * KNP * 2, tid);
  async_stage_b128(sW2, W2T, HID * HID * 2, tid);
  for (int i = tid; i < 4 * HID; i += 256) sAff[i] = aff[i];
  async_wait_all();

  const int lane = tid & 31, wave = tid >> 5;
  const int rt = wave & 3, cg = wave >> 2;
  const int mbase = rt * 16 + 8 * (lane >> 4);
  const int bn = lane & 15;
  const int numTiles = (N_NODES + 63) / 64;

  for (int tile = blockIdx.x; tile < numTiles; tile += gridDim.x) {
    const int n0 = tile * 64;
    {  // assemble A row = [h[n](69) | aggr(128) | 0...]
      const int r = tid >> 2, q = tid & 3;
      const int n = n0 + r;
      bf16* row = &sA1[r * KNP];
      if (n < N_NODES) {
        const bf16* hn = &hbf[(size_t)n * HSTR];
        const float* ms = &msg_sum[(size_t)n * HID];
        const float inv = 1.0f / fmaxf(cnt[n], 1.0f);
        if (q == 0) {
          sG[r] = graph_id[n];
          sC[r * 3 + 0] = coords[n * 3 + 0];
          sC[r * 3 + 1] = coords[n * 3 + 1];
          sC[r * 3 + 2] = coords[n * 3 + 2];
          for (int k = 0; k < 56; ++k) row[k] = hn[k];
        } else if (q == 1) {
          for (int k = 56; k < 69; ++k) row[k] = hn[k];
          for (int j = 0; j < 43; ++j) row[69 + j] = (bf16)(ms[j] * inv);
        } else if (q == 2) {
          for (int j = 43; j < 99; ++j) row[69 + j] = (bf16)(ms[j] * inv);
        } else {
          for (int j = 99; j < 128; ++j) row[69 + j] = (bf16)(ms[j] * inv);
          for (int k = KNR; k < KNP; ++k) row[k] = (bf16)0.0f;
        }
      } else {
        if (q == 0) { sG[r] = 0; sC[r*3+0] = sC[r*3+1] = sC[r*3+2] = 0.0f; }
        for (int k = q * 56; k < q * 56 + 56; ++k) row[k] = (bf16)0.0f;
      }
    }
    __syncthreads();

    v8f acc[4];
    gemm_tile<KNP, KNP / 32>(sA1, sW1, lane, rt, cg, acc);

#pragma unroll
    for (int t = 0; t < 4; ++t) {
      const int col = (cg * 4 + t) * 16 + bn;
      const float s1 = sAff[col], c1 = sAff[HID + col];
#pragma unroll
      for (int i = 0; i < 8; ++i) {
        float v = acc[t][i] * s1 + c1;
        v = v > 0.0f ? v : 0.01f * v;
        sA2[(mbase + i) * HID + col] = (bf16)v;
      }
    }
    __syncthreads();

    gemm_tile<HID, HID / 32>(sA2, sW2, lane, rt, cg, acc);

    // epilogue: h2 x coords outer product, segment-sum into pooled[G][384]
#pragma unroll
    for (int t = 0; t < 4; ++t) {
      const int col = (cg * 4 + t) * 16 + bn;
      const float s2 = sAff[2 * HID + col], c2 = sAff[3 * HID + col];
#pragma unroll
      for (int i = 0; i < 8; ++i) {
        const int r = mbase + i;
        if (n0 + r < N_NODES) {
          const float v = acc[t][i] * s2 + c2;
          float* pg = &pooled[(size_t)sG[r] * KZ + col * 3];
          atomicAdd(pg + 0, v * sC[r * 3 + 0]);
          atomicAdd(pg + 1, v * sC[r * 3 + 1]);
          atomicAdd(pg + 2, v * sC[r * 3 + 2]);
        }
      }
    }
    __syncthreads();
  }
}

// ---------------- kernel 6: mu/sigma heads ----------------
__global__ __launch_bounds__(256) void head_gemm(
    const float* __restrict__ pooled, const bf16* __restrict__ WmT,
    const bf16* __restrict__ WsT, const float* __restrict__ affMu,
    const float* __restrict__ affSg, float* __restrict__ out) {
  __shared__ bf16 sA[MP * KZ];
  const int sel = blockIdx.x;                 // 0 = mu, 1 = sigma
  const bf16* WT = sel ? WsT : WmT;
  const float* aff = sel ? affSg : affMu;
  const int tid = threadIdx.x;
  for (int i = tid; i < MP * KZ; i += 256) sA[i] = (bf16)pooled[i];
  __syncthreads();

  const int lane = tid & 31, wave = tid >> 5;
  const int khalf = (lane >> 4) * 8;
  const int bn = lane & 15;
  const int bkh = (lane >> 4) * 8;
  for (int tt = wave; tt < (MP / 16) * (ZD / 16); tt += 8) {
    const int rt = tt >> 4, ct = tt & 15;
    const int arow = rt * 16 + (lane & 15);
    const int coln = ct * 16 + bn;
    v8f acc = (v8f)0.0f;
    for (int kc = 0; kc < KZ / 32; ++kc) {
      const bf16* ap = &sA[arow * KZ + kc * 32 + khalf];
      const bf16* bp = &WT[(size_t)coln * KZ + kc * 32 + bkh];
      acc = wmma_bf16(frag2(ap, ap + 16), frag2(bp, bp + 16), acc);
    }
    const float s = aff[coln], c = aff[ZD + coln];
    const int mb = rt * 16 + 8 * (lane >> 4);
#pragma unroll
    for (int i = 0; i < 8; ++i) {
      const int row = mb + i;
      if (row < N_GRAPH) {
        out[(size_t)sel * N_GRAPH * ZD + (size_t)row * ZD + coln] =
            fmaxf(acc[i] * s + c, 0.0f);
      }
    }
  }
}

// ---------------- host launcher ----------------
// Input order: coords, mu_r_norm, edge_feat, params(jax pytree, dict keys sorted:
// edge[12], emb[16], mu[6], node[12], sigma[6]), node_cat, src, dst, graph_id.
extern "C" void kernel_launch(void* const* d_in, const int* in_sizes, int n_in,
                              void* d_out, int out_size, void* d_ws, size_t ws_size,
                              hipStream_t stream) {
  const float* coords    = (const float*)d_in[0];
  const float* mu_r      = (const float*)d_in[1];
  const float* edge_feat = (const float*)d_in[2];

  PrepArgs pa;
  pa.W1e = (const float*)d_in[3];  pa.b1e = (const float*)d_in[4];
  pa.g1e = (const float*)d_in[5];  pa.be1e = (const float*)d_in[6];
  pa.m1e = (const float*)d_in[7];  pa.v1e = (const float*)d_in[8];
  pa.W2e = (const float*)d_in[9];  pa.b2e = (const float*)d_in[10];
  pa.g2e = (const float*)d_in[11]; pa.be2e = (const float*)d_in[12];
  pa.m2e = (const float*)d_in[13]; pa.v2e = (const float*)d_in[14];
  EmbPtrs emb;
  for (int i = 0; i < 16; ++i) emb.p[i] = (const float*)d_in[15 + i];
  pa.Wm  = (const float*)d_in[31]; pa.bmu = (const float*)d_in[32];
  pa.gmu = (const float*)d_in[33]; pa.bemu = (const float*)d_in[34];
  pa.mmu = (const float*)d_in[35]; pa.vmu = (const float*)d_in[36];
  pa.W1n = (const float*)d_in[37]; pa.b1n = (const float*)d_in[38];
  pa.g1n = (const float*)d_in[39]; pa.be1n = (const float*)d_in[40];
  pa.m1n = (const float*)d_in[41]; pa.v1n = (const float*)d_in[42];
  pa.W2n = (const float*)d_in[43]; pa.b2n = (const float*)d_in[44];
  pa.g2n = (const float*)d_in[45]; pa.be2n = (const float*)d_in[46];
  pa.m2n = (const float*)d_in[47]; pa.v2n = (const float*)d_in[48];
  pa.Ws  = (const float*)d_in[49]; pa.bsg = (const float*)d_in[50];
  pa.gsg = (const float*)d_in[51]; pa.besg = (const float*)d_in[52];
  pa.msig = (const float*)d_in[53]; pa.vsig = (const float*)d_in[54];
  const int* node_cat = (const int*)d_in[55];
  const int* srci     = (const int*)d_in[56];
  const int* dsti     = (const int*)d_in[57];
  const int* graph_id = (const int*)d_in[58];

  // workspace carve-out
  char* ws = (char*)d_ws;
  size_t off = 0;
  auto carve = [&](size_t bytes) -> void* {
    void* p = ws + off;
    off = (off + bytes + 255) & ~(size_t)255;
    return p;
  };
  bf16*  hbf     = (bf16*)carve((size_t)N_NODES * HSTR * 2);
  float* msg_sum = (float*)carve((size_t)N_NODES * HID * 4);
  float* cnt     = (float*)carve((size_t)N_NODES * 4);
  float* pooled  = (float*)carve((size_t)MP * KZ * 4);
  pa.W1eT = (bf16*)carve(HID * K1P * 2);
  pa.W2eT = (bf16*)carve(HID * HID * 2);
  pa.W1nT = (bf16*)carve(HID * KNP * 2);
  pa.W2nT = (bf16*)carve(HID * HID * 2);
  pa.WmT  = (bf16*)carve(ZD * KZ * 2);
  pa.WsT  = (bf16*)carve(ZD * KZ * 2);
  pa.affE = (float*)carve(4 * HID * 4);
  pa.affN = (float*)carve(4 * HID * 4);
  pa.affMu = (float*)carve(2 * ZD * 4);
  pa.affSg = (float*)carve(2 * ZD * 4);

  zero_ws<<<2048, 256, 0, stream>>>(msg_sum, cnt, pooled);
  prep_weights<<<512, 256, 0, stream>>>(pa);
  atom_encode<<<(N_NODES + 3) / 4, 288, 0, stream>>>(emb, node_cat, mu_r, hbf);
  edge_mlp_scatter<<<1024, 256, 0, stream>>>(hbf, coords, edge_feat, srci, dsti,
                                             pa.W1eT, pa.W2eT, pa.affE,
                                             msg_sum, cnt);
  node_mlp_pool<<<512, 256, 0, stream>>>(hbf, msg_sum, cnt, coords, graph_id,
                                         pa.W1nT, pa.W2nT, pa.affN, pooled);
  head_gemm<<<2, 256, 0, stream>>>(pooled, pa.WmT, pa.WsT, pa.affMu, pa.affSg,
                                   (float*)d_out);
  (void)in_sizes; (void)n_in; (void)out_size; (void)ws_size;
}
